// RNNHighwayDecoder_56057913148029
// MI455X (gfx1250) — matrix operand order repaired and checked by reading, not compiled, not run on hardware
//
#include <hip/hip_runtime.h>
#include <hip/hip_bf16.h>
#include <math.h>

typedef __bf16 bf16_t;
typedef __attribute__((ext_vector_type(16))) __bf16 v16bf;
typedef __attribute__((ext_vector_type(8)))  __bf16 v8bf;
typedef __attribute__((ext_vector_type(8)))  float  v8f;

#define V_SZ 50257
#define E_SZ 512
#define H_SZ 512
#define B_SZ 32
#define S_SZ 60
#define T_SZ 48
#define NPAD 50272          // 3142 * 16
#define NT_TILES 3142
#define MROWS (T_SZ * B_SZ) // 1536
#define BS (B_SZ * S_SZ)    // 1920

// ---------- helpers ----------
__device__ __forceinline__ void split_f32(float x, bf16_t& hi, bf16_t& lo) {
    hi = (bf16_t)x;
    lo = (bf16_t)(x - (float)hi);
}

// A operand (M x K, row-major, K contiguous). base points at row 0 of the 16-row tile.
// A as loaded here (contiguous k, same order as B): lane<16 -> row=lane, K = [kk..kk+16)
//                  lane>=16 -> row=lane-16, K = [kk+16..kk+32)
__device__ __forceinline__ v16bf load_a_tile(const bf16_t* base, int lda, int lane, int kk) {
    int row = lane & 15;
    int c0  = kk + ((lane >> 4) << 4);
    const bf16_t* p = base + (size_t)row * lda + c0;
    v8bf p0 = *(const v8bf*)(p);
    v8bf p1 = *(const v8bf*)(p + 8);
    v16bf a;
#pragma unroll
    for (int i = 0; i < 8; ++i) { a[i] = p0[i]; a[i + 8] = p1[i]; }
    return a;
}

// B operand from transposed weight (N x K, row-major, K contiguous).
// B here (contiguous k, as in A): lane<16 -> col=lane, K=[kk..kk+16); lane>=16 -> col=lane-16, K=[kk+16..kk+32)
__device__ __forceinline__ v16bf load_b_tile(const bf16_t* baseNt, int ldb, int lane, int kk) {
    int n  = lane & 15;
    int k0 = kk + ((lane >> 4) << 4);
    return *(const v16bf*)(baseNt + (size_t)n * ldb + k0);
}

// error-compensated bf16 product: hi*hi + hi*lo + lo*hi  (~16 mantissa bits)
__device__ __forceinline__ v8f wmma3(v8f acc, v16bf ah, v16bf al, v16bf bh, v16bf bl) {
    acc = __builtin_amdgcn_wmma_f32_16x16x32_bf16(false, ah, false, bh, (short)0, acc, false, false);
    acc = __builtin_amdgcn_wmma_f32_16x16x32_bf16(false, ah, false, bl, (short)0, acc, false, false);
    acc = __builtin_amdgcn_wmma_f32_16x16x32_bf16(false, al, false, bh, (short)0, acc, false, false);
    return acc;
}

// ---------- prep kernels ----------
__global__ void k_transpose_split(const float* __restrict__ in, int R, int C, int Cpad,
                                  bf16_t* __restrict__ out_hi, bf16_t* __restrict__ out_lo) {
    __shared__ float tile[16][17];
    int c0 = blockIdx.x * 16, r0 = blockIdx.y * 16;
    int tx = threadIdx.x, ty = threadIdx.y;
    int r = r0 + ty, c = c0 + tx;
    float v = 0.f;
    if (r < R && c < C) v = in[(size_t)r * C + c];
    tile[ty][tx] = v;
    __syncthreads();
    int oc = c0 + ty;   // output row (= original column)
    int orr = r0 + tx;  // output col (= original row)
    if (oc < Cpad && orr < R) {
        float val = (oc < C) ? tile[tx][ty] : 0.f;
        bf16_t h, l; split_f32(val, h, l);
        out_hi[(size_t)oc * R + orr] = h;
        out_lo[(size_t)oc * R + orr] = l;
    }
}

__global__ void k_split(const float* __restrict__ in, bf16_t* __restrict__ hi,
                        bf16_t* __restrict__ lo, int n) {
    int i = blockIdx.x * blockDim.x + threadIdx.x;
    if (i < n) { bf16_t h, l; split_f32(in[i], h, l); hi[i] = h; lo[i] = l; }
}

__global__ void k_copy(const float* __restrict__ in, float* __restrict__ out, int n) {
    int i = blockIdx.x * blockDim.x + threadIdx.x;
    if (i < n) out[i] = in[i];
}

__global__ void k_gather_emb(const int* __restrict__ target, const float* __restrict__ embW,
                             bf16_t* __restrict__ emb_hi, bf16_t* __restrict__ emb_lo) {
    int row = blockIdx.x;                 // t*B + b
    int tok = target[row];
    const float* src = embW + (size_t)tok * E_SZ;
    for (int e = threadIdx.x; e < E_SZ; e += blockDim.x) {
        bf16_t h, l; split_f32(src[e], h, l);
        emb_hi[(size_t)row * E_SZ + e] = h;
        emb_lo[(size_t)row * E_SZ + e] = l;
    }
}

// ---------- energy = tanh(enc_h @ attn_W + b) : M=1920, K=1024, N=512 ----------
__global__ void k_energy(const bf16_t* __restrict__ A_hi, const bf16_t* __restrict__ A_lo,
                         const bf16_t* __restrict__ Bt_hi, const bf16_t* __restrict__ Bt_lo,
                         const float* __restrict__ bias, float* __restrict__ energy) {
    int wave = blockIdx.x * (blockDim.x >> 5) + (threadIdx.x >> 5);
    int lane = threadIdx.x & 31;
    const int NT = H_SZ / 16;
    int mt = wave / NT, nt = wave % NT;
    if (mt >= BS / 16) return;
    const bf16_t* Ah = A_hi + (size_t)mt * 16 * 1024;
    const bf16_t* Al = A_lo + (size_t)mt * 16 * 1024;
    const bf16_t* Bh = Bt_hi + (size_t)nt * 16 * 1024;
    const bf16_t* Bl = Bt_lo + (size_t)nt * 16 * 1024;
    v8f acc = {};
    for (int kk = 0; kk < 1024; kk += 32) {
        v16bf ah = load_a_tile(Ah, 1024, lane, kk);
        v16bf al = load_a_tile(Al, 1024, lane, kk);
        v16bf bh = load_b_tile(Bh, 1024, lane, kk);
        v16bf bl = load_b_tile(Bl, 1024, lane, kk);
        acc = wmma3(acc, ah, al, bh, bl);
    }
    int n = nt * 16 + (lane & 15);
    int rbase = mt * 16 + ((lane >> 4) << 3);
    float b = bias[n];
#pragma unroll
    for (int r = 0; r < 8; ++r)
        energy[(size_t)(rbase + r) * H_SZ + n] = tanhf(acc[r] + b);
}

// ---------- attention + softmax + context, per step (one block per batch) ----------
__global__ void k_attn(const float* __restrict__ enc_h, const float* __restrict__ energy,
                       const float* __restrict__ s, float* __restrict__ ctx_e,
                       bf16_t* __restrict__ g_hi, bf16_t* __restrict__ g_lo) {
    int b = blockIdx.x;
    __shared__ float sh_s[H_SZ];
    __shared__ float sc[S_SZ];
    int tid = threadIdx.x;
    // CDNA5 async global->LDS staging of the recurrent state
    for (int i = tid; i < H_SZ; i += blockDim.x) {
        unsigned lds_off = (unsigned)(uintptr_t)(&sh_s[i]);
        const float* gp = s + (size_t)b * H_SZ + i;
        asm volatile("global_load_async_to_lds_b32 %0, %1, off"
                     :: "v"(lds_off), "v"(gp) : "memory");
    }
    asm volatile("s_wait_asynccnt 0x0" ::: "memory");
    __syncthreads();
    int wv = tid >> 5, lane = tid & 31;
    for (int i = wv; i < S_SZ; i += 8) {
        const float* erow = energy + ((size_t)b * S_SZ + i) * H_SZ;
        float p = 0.f;
        for (int k = lane; k < H_SZ; k += 32) p += erow[k] * sh_s[k];
        for (int off = 16; off; off >>= 1) p += __shfl_xor(p, off, 32);
        if (lane == 0) sc[i] = p;
    }
    __syncthreads();
    if (tid == 0) {
        float m = sc[0];
        for (int i = 1; i < S_SZ; ++i) m = fmaxf(m, sc[i]);
        float ssum = 0.f;
        for (int i = 0; i < S_SZ; ++i) { sc[i] = expf(sc[i] - m); ssum += sc[i]; }
        float inv = 1.f / ssum;
        for (int i = 0; i < S_SZ; ++i) sc[i] *= inv;
    }
    __syncthreads();
    for (int d = tid; d < E_SZ; d += blockDim.x) {
        const float* eh = enc_h + (size_t)b * S_SZ * 1024 + d;
        float acc = 0.f;
        for (int i = 0; i < S_SZ; ++i) acc += sc[i] * eh[(size_t)i * 1024];
        ctx_e[(size_t)b * E_SZ + d] = acc;
        float gv = sh_s[d] * acc;
        bf16_t h, l; split_f32(gv, h, l);
        g_hi[(size_t)b * E_SZ + d] = h;
        g_lo[(size_t)b * E_SZ + d] = l;
    }
}

// ---------- one highway layer step: GEMM + fused gate ----------
__global__ void k_cell(int layer, int t,
                       const bf16_t* __restrict__ g_hi, const bf16_t* __restrict__ g_lo,
                       const bf16_t* __restrict__ cWt_hi, const bf16_t* __restrict__ cWt_lo,
                       const float* __restrict__ cell_b,
                       const bf16_t* __restrict__ emb_hi, const bf16_t* __restrict__ emb_lo,
                       const bf16_t* __restrict__ iWt_hi, const bf16_t* __restrict__ iWt_lo,
                       const float* __restrict__ inp_b,
                       const float* __restrict__ ctx_e, float* __restrict__ s,
                       bf16_t* __restrict__ g2_hi, bf16_t* __restrict__ g2_lo,
                       bf16_t* __restrict__ dec_hi, bf16_t* __restrict__ dec_lo) {
    __shared__ float wsmem[2][B_SZ][16];
    int tid = threadIdx.x, wv = tid >> 5, lane = tid & 31;
    int mt = wv & 1, half = wv >> 1;
    int c = blockIdx.x * 16;
    int nrow = half * 512 + c;
    const bf16_t* Bh = cWt_hi + (size_t)nrow * H_SZ;
    const bf16_t* Bl = cWt_lo + (size_t)nrow * H_SZ;
    const bf16_t* Ah = g_hi + (size_t)mt * 16 * E_SZ;
    const bf16_t* Al = g_lo + (size_t)mt * 16 * E_SZ;
    v8f acc = {};
    for (int kk = 0; kk < H_SZ; kk += 32) {
        v16bf ah = load_a_tile(Ah, E_SZ, lane, kk);
        v16bf al = load_a_tile(Al, E_SZ, lane, kk);
        v16bf bh = load_b_tile(Bh, H_SZ, lane, kk);
        v16bf bl = load_b_tile(Bl, H_SZ, lane, kk);
        acc = wmma3(acc, ah, al, bh, bl);
    }
    if (layer == 0) {
        const bf16_t* Xh = emb_hi + ((size_t)t * B_SZ + mt * 16) * E_SZ;
        const bf16_t* Xl = emb_lo + ((size_t)t * B_SZ + mt * 16) * E_SZ;
        const bf16_t* Ih = iWt_hi + (size_t)nrow * E_SZ;
        const bf16_t* Il = iWt_lo + (size_t)nrow * E_SZ;
        for (int kk = 0; kk < E_SZ; kk += 32) {
            v16bf xh = load_a_tile(Xh, E_SZ, lane, kk);
            v16bf xl = load_a_tile(Xl, E_SZ, lane, kk);
            v16bf ih = load_b_tile(Ih, E_SZ, lane, kk);
            v16bf il = load_b_tile(Il, E_SZ, lane, kk);
            acc = wmma3(acc, xh, xl, ih, il);
        }
    }
    int n = c + (lane & 15);
    float bias = cell_b[half * 512 + n];
    if (layer == 0) bias += inp_b[half * 512 + n];
    int rbase = mt * 16 + ((lane >> 4) << 3);
#pragma unroll
    for (int r = 0; r < 8; ++r)
        wsmem[half][rbase + r][lane & 15] = acc[r] + bias;
    __syncthreads();
    for (int idx = tid; idx < B_SZ * 16; idx += blockDim.x) {
        int m = idx >> 4, nl = idx & 15;
        int col = c + nl;
        float hh = tanhf(wsmem[0][m][nl]);
        float tt = 1.f / (1.f + expf(-wsmem[1][m][nl]));
        float sold = s[(size_t)m * H_SZ + col];
        float sn = hh * tt + sold * (1.f - tt);
        s[(size_t)m * H_SZ + col] = sn;
        bf16_t h, l;
        if (layer == 0) {
            float gv = sn * ctx_e[(size_t)m * E_SZ + col];
            split_f32(gv, h, l);
            g2_hi[(size_t)m * E_SZ + col] = h;
            g2_lo[(size_t)m * E_SZ + col] = l;
        } else {
            split_f32(sn, h, l);
            size_t o = ((size_t)t * B_SZ + m) * H_SZ + col;
            dec_hi[o] = h; dec_lo[o] = l;
        }
    }
}

// ---------- final projection: (1536 x 512) @ (512 x 50257) ----------
// wave: 4 m-tiles x 2 n-tiles; block's 4 waves share the n-tile pair (B dedups in WGP$)
__global__ void k_logits(const bf16_t* __restrict__ dec_hi, const bf16_t* __restrict__ dec_lo,
                         const bf16_t* __restrict__ Wt_hi, const bf16_t* __restrict__ Wt_lo,
                         const float* __restrict__ out_b, float* __restrict__ out) {
    int wid  = threadIdx.x >> 5;
    int lane = threadIdx.x & 31;
    int ng = blockIdx.x % (NT_TILES / 2);   // n-tile pair index (0..1570)
    int mq = blockIdx.x / (NT_TILES / 2);   // m-quarter (0..5), 16 m-tiles per block
    int nt0 = ng * 2;
    int mbase = (mq * 16 + wid * 4) * 16;   // first row of this wave's 4 m-tiles
    const bf16_t* B0h = Wt_hi + (size_t)nt0 * 16 * H_SZ;
    const bf16_t* B0l = Wt_lo + (size_t)nt0 * 16 * H_SZ;
    const bf16_t* B1h = B0h + (size_t)16 * H_SZ;
    const bf16_t* B1l = B0l + (size_t)16 * H_SZ;
    v8f acc[4][2] = {};
    for (int kk = 0; kk < H_SZ; kk += 32) {
        __builtin_prefetch(B0h + (size_t)(lane & 15) * H_SZ + kk + 64, 0, 0);
        v16bf b0h = load_b_tile(B0h, H_SZ, lane, kk);
        v16bf b0l = load_b_tile(B0l, H_SZ, lane, kk);
        v16bf b1h = load_b_tile(B1h, H_SZ, lane, kk);
        v16bf b1l = load_b_tile(B1l, H_SZ, lane, kk);
#pragma unroll
        for (int m = 0; m < 4; ++m) {
            const bf16_t* Ah = dec_hi + (size_t)(mbase + m * 16) * H_SZ;
            const bf16_t* Al = dec_lo + (size_t)(mbase + m * 16) * H_SZ;
            v16bf ah = load_a_tile(Ah, H_SZ, lane, kk);
            v16bf al = load_a_tile(Al, H_SZ, lane, kk);
            acc[m][0] = wmma3(acc[m][0], ah, al, b0h, b0l);
            acc[m][1] = wmma3(acc[m][1], ah, al, b1h, b1l);
        }
    }
#pragma unroll
    for (int j = 0; j < 2; ++j) {
        int n = (nt0 + j) * 16 + (lane & 15);
        if (n >= V_SZ) continue;
        float bias = out_b[n];
#pragma unroll
        for (int m = 0; m < 4; ++m) {
            int r0 = mbase + m * 16 + ((lane >> 4) << 3);
#pragma unroll
            for (int r = 0; r < 8; ++r)
                out[(size_t)(r0 + r) * V_SZ + n] = acc[m][j][r] + bias;
        }
    }
}

// ---------- host ----------
extern "C" void kernel_launch(void* const* d_in, const int* in_sizes, int n_in,
                              void* d_out, int out_size, void* d_ws, size_t ws_size,
                              hipStream_t stream) {
    const float* enc_h  = (const float*)d_in[0];
    const float* prev_s = (const float*)d_in[1];
    const int*   target = (const int*)d_in[2];
    const float* embW   = (const float*)d_in[3];
    const float* attnW  = (const float*)d_in[4];
    const float* attnb  = (const float*)d_in[5];
    const float* inpW   = (const float*)d_in[6];
    const float* inpb   = (const float*)d_in[7];
    const float* cellW  = (const float*)d_in[8];
    const float* cellb  = (const float*)d_in[9];
    const float* outW   = (const float*)d_in[10];
    const float* outb   = (const float*)d_in[11];
    float* out = (float*)d_out;

    char* ws = (char*)d_ws;
    size_t off = 0;
    auto alloc = [&](size_t bytes) -> void* {
        void* p = ws + off;
        off += (bytes + 255) & ~(size_t)255;
        return p;
    };

    bf16_t* Wt_hi   = (bf16_t*)alloc((size_t)NPAD * H_SZ * 2);
    bf16_t* Wt_lo   = (bf16_t*)alloc((size_t)NPAD * H_SZ * 2);
    bf16_t* ench_hi = (bf16_t*)alloc((size_t)BS * 1024 * 2);
    bf16_t* ench_lo = (bf16_t*)alloc((size_t)BS * 1024 * 2);
    bf16_t* aWt_hi  = (bf16_t*)alloc((size_t)H_SZ * 1024 * 2);
    bf16_t* aWt_lo  = (bf16_t*)alloc((size_t)H_SZ * 1024 * 2);
    bf16_t* cWt_hi  = (bf16_t*)alloc((size_t)2 * 1024 * H_SZ * 2);
    bf16_t* cWt_lo  = (bf16_t*)alloc((size_t)2 * 1024 * H_SZ * 2);
    bf16_t* iWt_hi  = (bf16_t*)alloc((size_t)1024 * E_SZ * 2);
    bf16_t* iWt_lo  = (bf16_t*)alloc((size_t)1024 * E_SZ * 2);
    bf16_t* emb_hi  = (bf16_t*)alloc((size_t)MROWS * E_SZ * 2);
    bf16_t* emb_lo  = (bf16_t*)alloc((size_t)MROWS * E_SZ * 2);
    float*  energy  = (float*)alloc((size_t)BS * H_SZ * 4);
    float*  sbuf    = (float*)alloc((size_t)B_SZ * H_SZ * 4);
    float*  ctxe    = (float*)alloc((size_t)B_SZ * E_SZ * 4);
    bf16_t* g0_hi   = (bf16_t*)alloc((size_t)B_SZ * E_SZ * 2);
    bf16_t* g0_lo   = (bf16_t*)alloc((size_t)B_SZ * E_SZ * 2);
    bf16_t* g1_hi   = (bf16_t*)alloc((size_t)B_SZ * E_SZ * 2);
    bf16_t* g1_lo   = (bf16_t*)alloc((size_t)B_SZ * E_SZ * 2);
    bf16_t* dec_hi  = (bf16_t*)alloc((size_t)MROWS * H_SZ * 2);
    bf16_t* dec_lo  = (bf16_t*)alloc((size_t)MROWS * H_SZ * 2);
    (void)ws_size; (void)in_sizes; (void)n_in; (void)out_size;

    dim3 tb(16, 16);
    k_transpose_split<<<dim3(NPAD / 16, H_SZ / 16), tb, 0, stream>>>(outW, H_SZ, V_SZ, NPAD, Wt_hi, Wt_lo);
    k_transpose_split<<<dim3(512 / 16, 1024 / 16), tb, 0, stream>>>(attnW, 1024, 512, 512, aWt_hi, aWt_lo);
    for (int l = 0; l < 2; ++l)
        k_transpose_split<<<dim3(1024 / 16, 512 / 16), tb, 0, stream>>>(
            cellW + (size_t)l * 512 * 1024, 512, 1024, 1024,
            cWt_hi + (size_t)l * 1024 * 512, cWt_lo + (size_t)l * 1024 * 512);
    k_transpose_split<<<dim3(1024 / 16, 512 / 16), tb, 0, stream>>>(inpW, 512, 1024, 1024, iWt_hi, iWt_lo);
    k_split<<<(BS * 1024 + 255) / 256, 256, 0, stream>>>(enc_h, ench_hi, ench_lo, BS * 1024);
    k_gather_emb<<<MROWS, 128, 0, stream>>>(target, embW, emb_hi, emb_lo);
    k_copy<<<(B_SZ * H_SZ + 255) / 256, 256, 0, stream>>>(prev_s, sbuf, B_SZ * H_SZ);
    k_energy<<<(120 * 32) / 4, 128, 0, stream>>>(ench_hi, ench_lo, aWt_hi, aWt_lo, attnb, energy);
    for (int t = 0; t < T_SZ; ++t) {
        k_attn<<<B_SZ, 256, 0, stream>>>(enc_h, energy, sbuf, ctxe, g0_hi, g0_lo);
        k_cell<<<32, 128, 0, stream>>>(0, t, g0_hi, g0_lo, cWt_hi, cWt_lo, cellb,
                                       emb_hi, emb_lo, iWt_hi, iWt_lo, inpb,
                                       ctxe, sbuf, g1_hi, g1_lo, dec_hi, dec_lo);
        k_cell<<<32, 128, 0, stream>>>(1, t, g1_hi, g1_lo,
                                       cWt_hi + (size_t)1024 * 512, cWt_lo + (size_t)1024 * 512,
                                       cellb + 1024, emb_hi, emb_lo, iWt_hi, iWt_lo, inpb,
                                       ctxe, sbuf, g0_hi, g0_lo, dec_hi, dec_lo);
    }
    // 6 m-quarters x 1571 n-tile pairs, 4 waves per block
    k_logits<<<6 * (NT_TILES / 2), 128, 0, stream>>>(dec_hi, dec_lo, Wt_hi, Wt_lo, outb, out);
}
